// GCNLayer_75488345194724
// MI455X (gfx1250) — compile-verified, matrix operand order checked by reading, not run on hardware
//
#include <hip/hip_runtime.h>
#include <hip/hip_bf16.h>

typedef __attribute__((ext_vector_type(16))) __bf16 v16bf;
typedef __attribute__((ext_vector_type(8)))  __bf16 v8bf;
typedef __attribute__((ext_vector_type(8)))  float  v8f;

#define GCN_N 100000
#define GCN_D 128
#define GCN_E 1600000

// ---------------------------------------------------------------------------
// Kernel 1: zero the aggregation buffer, convert W (128x128 f32) -> bf16.
// ---------------------------------------------------------------------------
__global__ void gcn_init(float* __restrict__ agg, const float* __restrict__ W,
                         __bf16* __restrict__ Wb) {
  const int idx = blockIdx.x * blockDim.x + threadIdx.x;
  if (idx < GCN_N * GCN_D) agg[idx] = 0.0f;
  if (idx < GCN_D * GCN_D) Wb[idx] = (__bf16)W[idx];
}

// ---------------------------------------------------------------------------
// Kernel 2: edge-parallel gather/scatter-add.
// One wave32 per edge: lane loads float4 of feature[src], scales by norm[src],
// atomically adds into agg[dst]. feature (51.2MB) + agg (51.2MB) fit in the
// 192MB L2, so these atomics/gathers resolve at L2 bandwidth.
// ---------------------------------------------------------------------------
__global__ void gcn_scatter(const float* __restrict__ feat,
                            const float* __restrict__ norm,
                            const int*   __restrict__ src,
                            const int*   __restrict__ dst,
                            float*       __restrict__ agg) {
  const unsigned tid  = blockIdx.x * blockDim.x + threadIdx.x;
  const unsigned e    = tid >> 5;          // wave32: one edge per wave
  if (e >= GCN_E) return;
  const unsigned lane = threadIdx.x & 31u;
  const int s = src[e];
  const int d = dst[e];
  const float nv = norm[s];
  const float4 f = *(const float4*)(feat + (size_t)s * GCN_D + lane * 4);
  float* p = agg + (size_t)d * GCN_D + lane * 4;
  unsafeAtomicAdd(p + 0, f.x * nv);
  unsafeAtomicAdd(p + 1, f.y * nv);
  unsafeAtomicAdd(p + 2, f.z * nv);
  unsafeAtomicAdd(p + 3, f.w * nv);
}

// ---------------------------------------------------------------------------
// Kernel 3: out = (agg * norm) @ W.T + b via v_wmma_f32_16x16x32_bf16.
// One wave computes a 16x128 output slab (8 accumulator tiles, K-loop 4x32).
//
// A operand (16x32 bf16, ISA 7.12.2): lane L holds row m = L%16, halves of the
// K-chunk: elements [half*8, half*8+8) in v[0..3], [16+half*8, ...) in v[4..7]
// -> two contiguous 8-element runs per lane. Dense B (32x16) mirrors A with
// lane -> column N; since B = W^T, column n of B is row n of W, so B loads are
// the same contiguous pattern from W's rows.
// C/D (16x16 f32): VGPR v / lane-half -> row v + 8*half, col = lane%16.
// ---------------------------------------------------------------------------
__global__ void gcn_gemm(const float*  __restrict__ agg,
                         const float*  __restrict__ norm,
                         const __bf16* __restrict__ Wb,
                         const float*  __restrict__ bias,
                         float*        __restrict__ out) {
  const int wave     = threadIdx.x >> 5;
  const int rowBlock = blockIdx.x * (blockDim.x >> 5) + wave;
  if (rowBlock >= GCN_N / 16) return;
  const int lane   = threadIdx.x & 31;
  const int lane16 = lane & 15;
  const int half   = lane >> 4;

  const int m = rowBlock * 16 + lane16;       // A row for this lane
  const float nm = norm[m];
  const float* arow = agg + (size_t)m * GCN_D;

  v8f acc[8];
#pragma unroll
  for (int t = 0; t < 8; ++t) acc[t] = (v8f){0, 0, 0, 0, 0, 0, 0, 0};

#pragma unroll
  for (int kb = 0; kb < 4; ++kb) {
    const int k0 = kb * 32 + half * 8;
    // Build A: (agg * norm) -> bf16, per-lane K runs [k0,k0+8) and [k0+16,k0+24)
    float4 f0 = *(const float4*)(arow + k0);
    float4 f1 = *(const float4*)(arow + k0 + 4);
    float4 f2 = *(const float4*)(arow + k0 + 16);
    float4 f3 = *(const float4*)(arow + k0 + 20);
    v16bf a;
    a[0]  = (__bf16)(f0.x * nm); a[1]  = (__bf16)(f0.y * nm);
    a[2]  = (__bf16)(f0.z * nm); a[3]  = (__bf16)(f0.w * nm);
    a[4]  = (__bf16)(f1.x * nm); a[5]  = (__bf16)(f1.y * nm);
    a[6]  = (__bf16)(f1.z * nm); a[7]  = (__bf16)(f1.w * nm);
    a[8]  = (__bf16)(f2.x * nm); a[9]  = (__bf16)(f2.y * nm);
    a[10] = (__bf16)(f2.z * nm); a[11] = (__bf16)(f2.w * nm);
    a[12] = (__bf16)(f3.x * nm); a[13] = (__bf16)(f3.y * nm);
    a[14] = (__bf16)(f3.z * nm); a[15] = (__bf16)(f3.w * nm);

#pragma unroll
    for (int t = 0; t < 8; ++t) {
      const __bf16* wrow = Wb + (size_t)(t * 16 + lane16) * GCN_D;  // B col n = W row n
      v8bf w0 = *(const v8bf*)(wrow + k0);
      v8bf w1 = *(const v8bf*)(wrow + k0 + 16);
      v16bf bm;
#pragma unroll
      for (int i = 0; i < 8; ++i) { bm[i] = w0[i]; bm[8 + i] = w1[i]; }
      acc[t] = __builtin_amdgcn_wmma_f32_16x16x32_bf16(
          /*neg_a=*/false, a, /*neg_b=*/false, bm,
          /*c_mod=*/(short)0, acc[t], /*reuse_a=*/false, /*reuse_b=*/false);
    }
  }

  // Epilogue: D layout -> row = base + v + 8*half, col = t*16 + lane16
#pragma unroll
  for (int t = 0; t < 8; ++t) {
    const int col = t * 16 + lane16;
    const float bv = bias[col];
#pragma unroll
    for (int v = 0; v < 8; ++v) {
      const int row = rowBlock * 16 + v + 8 * half;
      out[(size_t)row * GCN_D + col] = acc[t][v] + bv;
    }
  }
}

// ---------------------------------------------------------------------------
extern "C" void kernel_launch(void* const* d_in, const int* in_sizes, int n_in,
                              void* d_out, int out_size, void* d_ws, size_t ws_size,
                              hipStream_t stream) {
  const float* feature = (const float*)d_in[0];
  const float* norm    = (const float*)d_in[1];
  const int*   src     = (const int*)d_in[2];
  const int*   dst     = (const int*)d_in[3];
  const float* W       = (const float*)d_in[4];
  const float* b       = (const float*)d_in[5];
  float* out = (float*)d_out;

  float*  agg = (float*)d_ws;                                   // 51.2 MB
  __bf16* Wb  = (__bf16*)((char*)d_ws + (size_t)GCN_N * GCN_D * sizeof(float));

  // 1) zero agg + convert W to bf16
  const int nInit = GCN_N * GCN_D;
  gcn_init<<<(nInit + 255) / 256, 256, 0, stream>>>(agg, W, Wb);

  // 2) edge scatter-add: 1 wave32 per edge, 8 edges per 256-thread block
  gcn_scatter<<<GCN_E / 8, 256, 0, stream>>>(feature, norm, src, dst, agg);

  // 3) WMMA GEMM: 6250 row-blocks, 8 waves per block
  const int rowBlocks = GCN_N / 16;
  gcn_gemm<<<(rowBlocks + 7) / 8, 256, 0, stream>>>(agg, norm, Wb, b, out);
}
